// CrossNetworkMix_87162066305256
// MI455X (gfx1250) — matrix operand order repaired and checked by reading, not compiled, not run on hardware
//
#include <hip/hip_runtime.h>
#include <hip/hip_bf16.h>

// Problem constants (match reference)
#define B_ROWS   16384
#define D_DIM    512
#define L_LAYERS 3
#define R_RANK   64
#define E_EXP    4
#define ER_DIM   (E_EXP * R_RANK)      // 256
#define BM       16                    // rows per workgroup tile

// LDS row strides (halfs), padded so 16 rows hit distinct bank groups
#define XB_STRIDE 520                  // 512 + 8
#define VX_STRIDE 264                  // 256 + 8

typedef __attribute__((ext_vector_type(16))) __bf16       v16bf;
typedef __attribute__((ext_vector_type(8)))  float        v8f;
typedef __attribute__((ext_vector_type(4)))  unsigned int uint4v;
typedef __attribute__((ext_vector_type(8)))  unsigned int uint8v;

// f32 -> bf16 bits with round-to-nearest-even (storage type: unsigned short)
__device__ __forceinline__ unsigned short tobf(float f) {
  unsigned u = __float_as_uint(f);
  unsigned r = (u + 0x7FFFu + ((u >> 16) & 1u)) >> 16;
  return (unsigned short)r;
}

// Branch-free tanh: native V_TANH_F32 when the builtin exists, otherwise the
// exp2 identity tanh(x) = 1 - 2/(exp2(2*log2e*x) + 1) (monotone, saturates
// correctly at +-1, no libm branches).
__device__ __forceinline__ float fast_tanh(float x) {
#if __has_builtin(__builtin_amdgcn_tanhf)
  return __builtin_amdgcn_tanhf(x);
#elif __has_builtin(__builtin_amdgcn_tanh_f32)
  return __builtin_amdgcn_tanh_f32(x);
#else
  float t = __builtin_amdgcn_exp2f(x * 2.8853900817779268f); // 2*log2(e)
  return 1.0f - 2.0f * __builtin_amdgcn_rcpf(t + 1.0f);
#endif
}

#define LOG2E 1.4426950408889634f

// A-matrix 16x32 bf16 fragment: lane holds 16 contiguous K halfs.
// p = row_base + kb + (lane>>4)*16
__device__ __forceinline__ v16bf load_a_frag(const unsigned short* p) {
  uint4v lo = *(const uint4v*)(p);
  uint4v hi = *(const uint4v*)(p + 8);
  uint8v u = {lo[0], lo[1], lo[2], lo[3], hi[0], hi[1], hi[2], hi[3]};
  return __builtin_bit_cast(v16bf, u);
}

// B-matrix 32x16 bf16 fragment: lane holds 16 contiguous K halfs.
// p = col_base + kb + (lane>>4)*16
__device__ __forceinline__ v16bf load_b_frag(const unsigned short* p) {
  uint4v lo = *(const uint4v*)(p);
  uint4v hi = *(const uint4v*)(p + 8);
  uint8v u = {lo[0], lo[1], lo[2], lo[3], hi[0], hi[1], hi[2], hi[3]};
  return __builtin_bit_cast(v16bf, u);
}

__device__ __forceinline__ v8f wmma_bf16(v16bf a, v16bf b, v8f c) {
  return __builtin_amdgcn_wmma_f32_16x16x32_bf16(false, a, false, b, (short)0, c,
                                                 false, false);
}

// ---------------- weight repack: fp32 -> bf16 in WMMA B-operand layouts ----------------
// Vt[l][er][d]  = V[l][e][d][r]      (er = e*R + r)  : B for  X[16,512] x Vall[512,256]
// Ut[l][d][er]  = U[l][e][d][r]                      : B for  CVg[16,256] x Uall[256,512]
// Ct[l][e][rc][s] = C[l][e][s][rc]                   : B for  Vx_e[16,64] x C_e[64,64]
// Gt[c][d]      = c<E ? gate_w[c][d] : 0   (16 padded columns for the gate logits tile)
#define VT_ELEMS (L_LAYERS * ER_DIM * D_DIM)
#define UT_ELEMS (L_LAYERS * D_DIM * ER_DIM)
#define CT_ELEMS (L_LAYERS * E_EXP * R_RANK * R_RANK)
#define GT_ELEMS (16 * D_DIM)
#define PACK_TOTAL (VT_ELEMS + UT_ELEMS + CT_ELEMS + GT_ELEMS)

__global__ void pack_weights_kernel(const float* __restrict__ U,
                                    const float* __restrict__ V,
                                    const float* __restrict__ C,
                                    const float* __restrict__ GW,
                                    unsigned short* __restrict__ Vt,
                                    unsigned short* __restrict__ Ut,
                                    unsigned short* __restrict__ Ct,
                                    unsigned short* __restrict__ Gt) {
  int i = blockIdx.x * blockDim.x + threadIdx.x;
  if (i < VT_ELEMS) {
    int l = i / (ER_DIM * D_DIM);
    int rem = i % (ER_DIM * D_DIM);
    int er = rem / D_DIM, d = rem % D_DIM;
    int e = er / R_RANK, r = er % R_RANK;
    Vt[i] = tobf(V[(((long)l * E_EXP + e) * D_DIM + d) * R_RANK + r]);
    return;
  }
  i -= VT_ELEMS;
  if (i < UT_ELEMS) {
    int l = i / (D_DIM * ER_DIM);
    int rem = i % (D_DIM * ER_DIM);
    int d = rem / ER_DIM, er = rem % ER_DIM;
    int e = er / R_RANK, r = er % R_RANK;
    Ut[i] = tobf(U[(((long)l * E_EXP + e) * D_DIM + d) * R_RANK + r]);
    return;
  }
  i -= UT_ELEMS;
  if (i < CT_ELEMS) {
    int l = i / (E_EXP * R_RANK * R_RANK);
    int rem = i % (E_EXP * R_RANK * R_RANK);
    int e = rem / (R_RANK * R_RANK);
    int rem2 = rem % (R_RANK * R_RANK);
    int rc = rem2 / R_RANK, s = rem2 % R_RANK;
    Ct[i] = tobf(C[(((long)l * E_EXP + e) * R_RANK + s) * R_RANK + rc]);
    return;
  }
  i -= CT_ELEMS;
  if (i < GT_ELEMS) {
    int c = i / D_DIM, d = i % D_DIM;
    Gt[i] = (c < E_EXP) ? tobf(GW[c * D_DIM + d]) : (unsigned short)0;
  }
}

// ---------------- main fused kernel: 16-row tile through all 3 layers ----------------
__global__ __launch_bounds__(256, 1) void crossnet_moe_kernel(
    const float* __restrict__ x_in,
    const float* __restrict__ bias,
    const unsigned short* __restrict__ Vt,
    const unsigned short* __restrict__ Ut,
    const unsigned short* __restrict__ Ct,
    const unsigned short* __restrict__ Gt,
    float* __restrict__ out) {
  __shared__ float          lds_x0[BM * D_DIM];        // 32 KB: x0 tile (fp32)
  __shared__ float          lds_xl[BM * D_DIM];        // 32 KB: running xl (fp32)
  __shared__ unsigned short lds_xb[BM * XB_STRIDE];    // bf16 snapshot of xl (A operand)
  __shared__ unsigned short lds_vx[BM * VX_STRIDE];    // bf16 Vx tile [16 x 256]
  __shared__ unsigned short lds_cv[BM * VX_STRIDE];    // bf16 gated CVx tile [16 x 256]
  __shared__ float          lds_logit[BM * 16];        // full 16x16 logits tile (cols 0..3 valid)
  __shared__ float          lds_gate[BM * E_EXP];

  const int tid  = threadIdx.x;
  const int wave = tid >> 5;        // 8 waves (wave32)
  const int lane = tid & 31;
  const int nlo  = lane & 15;       // column / row-within-16
  const int nhi  = lane >> 4;       // half-wave selector
  const long tile_base = (long)blockIdx.x * BM * D_DIM;

  for (int i = tid; i < BM * D_DIM; i += 256) {
    float v = x_in[tile_base + i];
    lds_x0[i] = v;
    lds_xl[i] = v;
  }
  __syncthreads();

  for (int l = 0; l < L_LAYERS; ++l) {
    // bf16 snapshot of xl for the A operand
    for (int i = tid; i < BM * D_DIM; i += 256) {
      int r = i >> 9, c = i & 511;
      lds_xb[r * XB_STRIDE + c] = tobf(lds_xl[i]);
    }
    __syncthreads();

    // ---- Stage 1: Vx = tanh(X @ Vall) [16x256]; wave0 also computes gate logits ----
    const unsigned short* VtL = Vt + (long)l * ER_DIM * D_DIM;
    const int njobs = (wave == 0) ? 3 : 2;
    for (int j = 0; j < njobs; ++j) {
      const int job = (j == 2) ? 16 : (wave + j * 8);  // 0..15 = Vx tiles, 16 = gate
      const unsigned short* Brow = (job < 16)
          ? (VtL + (long)(job * 16 + nlo) * D_DIM)
          : (Gt + (long)nlo * D_DIM);
      const unsigned short* Arow = lds_xb + nlo * XB_STRIDE;
      v8f acc = {};
#pragma unroll 4
      for (int kb = 0; kb < D_DIM; kb += 32) {
        v16bf a = load_a_frag(Arow + kb + nhi * 16);
        v16bf b = load_b_frag(Brow + kb + nhi * 16);
        acc = wmma_bf16(a, b, acc);
      }
      if (job < 16) {
#pragma unroll
        for (int v = 0; v < 8; ++v) {
          int r = v + nhi * 8;
          lds_vx[r * VX_STRIDE + job * 16 + nlo] = tobf(fast_tanh(acc[v]));
        }
      } else {
        // uniform store of the whole padded logits tile (no lane divergence)
#pragma unroll
        for (int v = 0; v < 8; ++v) {
          int r = v + nhi * 8;
          lds_logit[r * 16 + nlo] = acc[v];
        }
      }
    }
    __syncthreads();

    // ---- gate softmax per row (softmax weights sum to 1 -> bias term simplifies) ----
    if (tid < BM) {
      float x0 = lds_logit[tid * 16 + 0];
      float x1 = lds_logit[tid * 16 + 1];
      float x2 = lds_logit[tid * 16 + 2];
      float x3 = lds_logit[tid * 16 + 3];
      float m = fmaxf(fmaxf(x0, x1), fmaxf(x2, x3));
      float e0 = __builtin_amdgcn_exp2f((x0 - m) * LOG2E);
      float e1 = __builtin_amdgcn_exp2f((x1 - m) * LOG2E);
      float e2 = __builtin_amdgcn_exp2f((x2 - m) * LOG2E);
      float e3 = __builtin_amdgcn_exp2f((x3 - m) * LOG2E);
      float inv = __builtin_amdgcn_rcpf(e0 + e1 + e2 + e3);
      lds_gate[tid * 4 + 0] = e0 * inv;
      lds_gate[tid * 4 + 1] = e1 * inv;
      lds_gate[tid * 4 + 2] = e2 * inv;
      lds_gate[tid * 4 + 3] = e3 * inv;
    }
    __syncthreads();

    // ---- Stage 2: CVg = gate_e * tanh(Vx_e @ C_e), block-diagonal [16x256] ----
    const unsigned short* CtL = Ct + (long)l * E_EXP * R_RANK * R_RANK;
    for (int j = 0; j < 2; ++j) {
      const int job = wave + j * 8;     // 0..15
      const int e = job >> 2;
      const int nt = job & 3;
      const unsigned short* Brow = CtL + ((long)e * R_RANK + nt * 16 + nlo) * R_RANK;
      const unsigned short* Arow = lds_vx + nlo * VX_STRIDE + e * R_RANK;
      v8f acc = {};
#pragma unroll
      for (int kb = 0; kb < R_RANK; kb += 32) {
        v16bf a = load_a_frag(Arow + kb + nhi * 16);
        v16bf b = load_b_frag(Brow + kb + nhi * 16);
        acc = wmma_bf16(a, b, acc);
      }
#pragma unroll
      for (int v = 0; v < 8; ++v) {
        int r = v + nhi * 8;
        float g = lds_gate[r * E_EXP + e];
        lds_cv[r * VX_STRIDE + e * R_RANK + nt * 16 + nlo] = tobf(g * fast_tanh(acc[v]));
      }
    }
    __syncthreads();

    // ---- Stage 3: G = CVg @ Uall [16x512]; xl += x0 * (bias + G) ----
    const unsigned short* UtL = Ut + (long)l * D_DIM * ER_DIM;
    const float* biasL = bias + l * D_DIM;
    for (int j = 0; j < 4; ++j) {
      const int dt = wave + j * 8;      // 0..31 column tiles
      const unsigned short* Brow = UtL + (long)(dt * 16 + nlo) * ER_DIM;
      const unsigned short* Arow = lds_cv + nlo * VX_STRIDE;
      v8f acc = {};
#pragma unroll 4
      for (int kb = 0; kb < ER_DIM; kb += 32) {
        v16bf a = load_a_frag(Arow + kb + nhi * 16);
        v16bf b = load_b_frag(Brow + kb + nhi * 16);
        acc = wmma_bf16(a, b, acc);
      }
      const int d = dt * 16 + nlo;
      const float bv = biasL[d];
#pragma unroll
      for (int v = 0; v < 8; ++v) {
        int r = v + nhi * 8;
        int idx = r * D_DIM + d;
        lds_xl[idx] += lds_x0[idx] * (bv + acc[v]);
      }
    }
    __syncthreads();
  }

  for (int i = tid; i < BM * D_DIM; i += 256) {
    out[tile_base + i] = lds_xl[i];
  }
}

extern "C" void kernel_launch(void* const* d_in, const int* in_sizes, int n_in,
                              void* d_out, int out_size, void* d_ws, size_t ws_size,
                              hipStream_t stream) {
  (void)in_sizes; (void)n_in; (void)out_size; (void)ws_size;
  const float* x_in = (const float*)d_in[0];
  const float* U    = (const float*)d_in[1];
  const float* V    = (const float*)d_in[2];
  const float* C    = (const float*)d_in[3];
  const float* bias = (const float*)d_in[4];
  const float* GW   = (const float*)d_in[5];
  float* out = (float*)d_out;

  unsigned short* ws = (unsigned short*)d_ws;
  unsigned short* Vt = ws;
  unsigned short* Ut = Vt + VT_ELEMS;
  unsigned short* Ct = Ut + UT_ELEMS;
  unsigned short* Gt = Ct + CT_ELEMS;

  pack_weights_kernel<<<(PACK_TOTAL + 255) / 256, 256, 0, stream>>>(
      U, V, C, GW, Vt, Ut, Ct, Gt);
  crossnet_moe_kernel<<<B_ROWS / BM, 256, 0, stream>>>(
      x_in, bias, Vt, Ut, Ct, Gt, out);
}
